// TaskSpecificSharedModel_13374528160159
// MI455X (gfx1250) — matrix operand order repaired and checked by reading, not compiled, not run on hardware
//
#include <hip/hip_runtime.h>
#include <hip/hip_bf16.h>

typedef _Float16 h16;
typedef __attribute__((ext_vector_type(16))) _Float16 v16h;
typedef __attribute__((ext_vector_type(8)))  _Float16 v8h;
typedef __attribute__((ext_vector_type(8)))  float    v8f;

#define NB 128      // batch
#define TT 50       // target len
#define TC 200      // claim len
#define HH 256      // GRU hidden
#define DD 512      // 2*H
#define EE 300      // emb dim
#define EP 320      // padded emb dim (mult of 32)
#define G3 768      // 3*H
#define NNODE 256   // padded node rows (250 real)
#define NREAL 250

__device__ __forceinline__ float sigm(float x) { return 1.0f / (1.0f + expf(-x)); }

// ---------------------------------------------------------------------------
// Generic f16 WMMA GEMM:  C[m][n] = sum_k A[m][k] * Bt[n][k]   (i.e. A @ Bt^T)
// Block = 4 waves in a 2(M) x 2(N) layout; each wave owns TWO 16x16 C tiles
// (adjacent in N) so A fragments feed 2 WMMAs. Block covers 32(M) x 64(N).
// MODE 0: C(f32)[m*N+n] = acc (+bias[n])
// MODE 1: C2(f16)[n*M+m] = acc   (transposed f16 store, vectorized v8h)
// MODE 2: v = relu(C[m*N+n] + acc); C[...] = v; C2(f16)[m*N+n] = v  (in place)
// ---------------------------------------------------------------------------
template<int MODE>
__global__ void __launch_bounds__(128) gemm16(
    const h16* __restrict__ A, size_t sA,
    const h16* __restrict__ Bt, size_t sB,
    float* __restrict__ C, h16* __restrict__ C2, size_t sC,
    const float* __restrict__ bias,
    int M, int N, int K, int ldA, int ldB)
{
    const int z    = blockIdx.z;
    const int wave = threadIdx.x >> 5;
    const int lane = threadIdx.x & 31;
    const int g    = lane >> 4;
    const int r    = lane & 15;
    const int wm   = wave & 1;
    const int wn   = wave >> 1;
    const int tm   = (blockIdx.x * 2 + wm) * 16;
    const int tn0  = (blockIdx.y * 4 + wn * 2) * 16;   // this wave: tn0, tn0+16

    const h16* Ab  = A  + (size_t)z * sA + (size_t)(tm  + r) * ldA + g * 16;
    const h16* Bb0 = Bt + (size_t)z * sB + (size_t)(tn0 + r) * ldB + g * 16;
    const h16* Bb1 = Bb0 + (size_t)16 * ldB;

    v8f acc[2];
#pragma unroll
    for (int j = 0; j < 8; ++j) { acc[0][j] = 0.0f; acc[1][j] = 0.0f; }

    for (int k0 = 0; k0 < K; k0 += 32) {
        v8h alo = *(const v8h*)(Ab + k0);
        v8h ahi = *(const v8h*)(Ab + k0 + 8);
        v16h a;
#pragma unroll
        for (int i = 0; i < 8; ++i) { a[i] = alo[i]; a[i + 8] = ahi[i]; }
        v16h b0 = *(const v16h*)(Bb0 + k0);
        v16h b1 = *(const v16h*)(Bb1 + k0);
        acc[0] = __builtin_amdgcn_wmma_f32_16x16x32_f16(
            false, a, false, b0, (short)0, acc[0], false, false);
        acc[1] = __builtin_amdgcn_wmma_f32_16x16x32_f16(
            false, a, false, b1, (short)0, acc[1], false, false);
    }

    if (MODE == 0) {
        float* Cb = C + (size_t)z * sC;
#pragma unroll
        for (int tt = 0; tt < 2; ++tt) {
            const int tn = tn0 + 16 * tt;
            float bv = bias ? bias[tn + r] : 0.0f;
#pragma unroll
            for (int j = 0; j < 8; ++j)
                Cb[(size_t)(tm + j + 8 * g) * N + tn + r] = acc[tt][j] + bv;
        }
    } else if (MODE == 1) {
        h16* C2b = C2 + (size_t)z * sC;
#pragma unroll
        for (int tt = 0; tt < 2; ++tt) {
            v8h o;
#pragma unroll
            for (int j = 0; j < 8; ++j) o[j] = (h16)acc[tt][j];
            *(v8h*)(C2b + (size_t)(tn0 + 16 * tt + r) * M + tm + 8 * g) = o;
        }
    } else {
        float* Cb = C + (size_t)z * sC;
        h16*  C2b = C2 + (size_t)z * sC;
#pragma unroll
        for (int tt = 0; tt < 2; ++tt) {
#pragma unroll
            for (int j = 0; j < 8; ++j) {
                size_t off = (size_t)(tm + j + 8 * g) * N + tn0 + 16 * tt + r;
                float v = acc[tt][j] + Cb[off];
                v = fmaxf(v, 0.0f);
                Cb[off]  = v;
                C2b[off] = (h16)v;
            }
        }
    }
}

// ---------------------------------------------------------------------------
// Persistent bidirectional-GRU passes, up to 8 independent (GRU x direction)
// passes fused into one launch (blockIdx.y selects the pass). Each WG owns
// 16 batch rows, keeps h (16x256 f16) in LDS (stride-padded), loops over all
// timesteps. Wave w computes the three gate tiles (cols w*16 of r/z/n),
// K=256 via 24 WMMAs. The 24 Whh B-fragments are MANUALLY hoisted into a
// fully-unrolled register array before the time loop (192 VGPRs resident),
// so the steady state is ds_load + v_wmma only.
// ---------------------------------------------------------------------------
struct GruPass {
    const float* gi;    // [B*T][768] precomputed x@Wih^T + bih
    const h16*  whh;    // [768][256] f16
    const float* bhh;   // [768]
    h16*        hs;     // [B][T][512]
    int T;
    int backward;
    int dirOff;
    int pad;
};
struct GruPack { GruPass p[8]; };

__global__ void __launch_bounds__(512) gru_seq_multi(GruPack pk)
{
    const GruPass P = pk.p[blockIdx.y];
    const float* __restrict__ gi   = P.gi;
    const h16*   __restrict__ whh  = P.whh;
    const float* __restrict__ bhh  = P.bhh;
    h16*         __restrict__ hout = P.hs;
    const int T        = P.T;
    const int backward = P.backward;
    const int dirOff   = P.dirOff;

    __shared__ __align__(16) h16 hsm[16 * 264];   // padded stride: no bank conflicts
    const int tid  = threadIdx.x;
    const int wave = tid >> 5;
    const int lane = tid & 31;
    const int g    = lane >> 4;
    const int r    = lane & 15;
    const int b0   = blockIdx.x * 16;

    for (int i = tid; i < 16 * 264; i += 512) hsm[i] = (h16)0.0f;
    __syncthreads();

    const h16* ar = hsm + r * 264 + g * 16;
    const int  n  = wave * 16 + r;
    const float br = bhh[n], bz = bhh[n + 256], bn = bhh[n + 512];
    const h16* Bbase = whh + (size_t)n * HH + g * 16;

    // Manually hoist all 24 Whh fragments into registers (constant-indexed,
    // fully unrolled -> promoted; immune to alias-analysis failures).
    v16h wf[24];
#pragma unroll
    for (int kk = 0; kk < 8; ++kk) {
        wf[kk * 3 + 0] = *(const v16h*)(Bbase + kk * 32);
        wf[kk * 3 + 1] = *(const v16h*)(Bbase + 256 * HH + kk * 32);
        wf[kk * 3 + 2] = *(const v16h*)(Bbase + 512 * HH + kk * 32);
    }

    for (int s = 0; s < T; ++s) {
        const int t = backward ? (T - 1 - s) : s;
        v8f cr, cz, cn;
#pragma unroll
        for (int j = 0; j < 8; ++j) { cr[j] = 0.f; cz[j] = 0.f; cn[j] = 0.f; }

#pragma unroll
        for (int kk = 0; kk < 8; ++kk) {
            v8h alo = *(const v8h*)(ar + kk * 32);        // LDS
            v8h ahi = *(const v8h*)(ar + kk * 32 + 8);    // LDS
            v16h a;
#pragma unroll
            for (int i = 0; i < 8; ++i) { a[i] = alo[i]; a[i + 8] = ahi[i]; }
            cr = __builtin_amdgcn_wmma_f32_16x16x32_f16(false, a, false, wf[kk * 3 + 0], (short)0, cr, false, false);
            cz = __builtin_amdgcn_wmma_f32_16x16x32_f16(false, a, false, wf[kk * 3 + 1], (short)0, cz, false, false);
            cn = __builtin_amdgcn_wmma_f32_16x16x32_f16(false, a, false, wf[kk * 3 + 2], (short)0, cn, false, false);
        }
        __syncthreads();  // all waves finished reading h

#pragma unroll
        for (int j = 0; j < 8; ++j) {
            const int m = j + 8 * g;
            const float* gib = gi + ((size_t)(b0 + m) * T + t) * G3;
            float rg = sigm(gib[n]       + cr[j] + br);
            float zg = sigm(gib[n + 256] + cz[j] + bz);
            float ng = tanhf(gib[n + 512] + rg * (cn[j] + bn));
            float hold = (float)hsm[m * 264 + n];
            float hnew = (1.0f - zg) * ng + zg * hold;
            hsm[m * 264 + n] = (h16)hnew;
            hout[((size_t)(b0 + m) * T + t) * DD + dirOff + n] = (h16)hnew;
        }
        __syncthreads();  // h fully updated before next step's reads
    }
}

// --------------------------- small helper kernels ---------------------------

__global__ void embed_gather(h16* __restrict__ out, const int* __restrict__ idx,
                             const float* __restrict__ emb, int rows)
{
    int i = blockIdx.x * 256 + threadIdx.x;
    if (i >= rows * EP) return;
    int row = i / EP, col = i - row * EP;
    float v = 0.0f;
    if (col < EE) v = emb[(size_t)idx[row] * EE + col];
    out[i] = (h16)v;
}

__global__ void cast_pad(h16* __restrict__ out, const float* __restrict__ in,
                         int N, int Kin, int Kpad)
{
    int i = blockIdx.x * 256 + threadIdx.x;
    if (i >= N * Kpad) return;
    int row = i / Kpad, col = i - row * Kpad;
    out[i] = (h16)(col < Kin ? in[(size_t)row * Kin + col] : 0.0f);
}

__global__ void node_init(h16* __restrict__ nf16, float* __restrict__ nf32,
                          const h16* __restrict__ tgt_hs, const h16* __restrict__ clm_hs)
{
    size_t i = (size_t)blockIdx.x * 256 + threadIdx.x;
    if (i >= (size_t)NB * NNODE * DD) return;
    int d = (int)(i & (DD - 1));
    size_t rest = i >> 9;
    int row = (int)(rest & (NNODE - 1));
    int b   = (int)(rest >> 8);
    h16 v = (h16)0.0f;
    if (row < TT)         v = tgt_hs[((size_t)b * TT + row) * DD + d];
    else if (row < NREAL) v = clm_hs[((size_t)b * TC + (row - TT)) * DD + d];
    nf16[i] = v;
    nf32[i] = (float)v;
}

__global__ void attn_softmax(const h16* __restrict__ tgt_hs,
                             const h16* __restrict__ clm_hs,
                             float* __restrict__ w)
{
    const int b = blockIdx.x, tid = threadIdx.x;   // 256 threads
    __shared__ float sh_t[DD];
    __shared__ float red[256];
    for (int i = tid; i < DD; i += 256)
        sh_t[i] = (float)tgt_hs[((size_t)b * TT + (TT - 1)) * DD + i];
    __syncthreads();
    float logit = -1e30f;
    if (tid < TC) {
        const h16* rowp = clm_hs + ((size_t)b * TC + tid) * DD;
        float acc = 0.0f;
        for (int d = 0; d < DD; ++d) acc += (float)rowp[d] * sh_t[d];
        logit = acc;
    }
    red[tid] = logit; __syncthreads();
    for (int s = 128; s > 0; s >>= 1) { if (tid < s) red[tid] = fmaxf(red[tid], red[tid + s]); __syncthreads(); }
    float mx = red[0]; __syncthreads();
    float e = (tid < TC) ? expf(logit - mx) : 0.0f;
    red[tid] = e; __syncthreads();
    for (int s = 128; s > 0; s >>= 1) { if (tid < s) red[tid] += red[tid + s]; __syncthreads(); }
    if (tid < TC) w[(size_t)b * TC + tid] = e / red[0];
}

__global__ void softmax_corr(const float* __restrict__ L, h16* __restrict__ O)
{
    const int i = blockIdx.x, b = blockIdx.y, tid = threadIdx.x;   // 256 threads
    size_t base = ((size_t)b * NNODE + i) * NNODE;
    if (i >= NREAL) { O[base + tid] = (h16)0.0f; return; }
    const bool act = tid < NREAL;
    float lv = act ? L[base + tid] : -1e30f;
    __shared__ float red[256];
    red[tid] = lv; __syncthreads();
    for (int s = 128; s > 0; s >>= 1) { if (tid < s) red[tid] = fmaxf(red[tid], red[tid + s]); __syncthreads(); }
    float mx = red[0]; __syncthreads();
    float e = act ? expf(lv - mx) : 0.0f;
    red[tid] = e; __syncthreads();
    for (int s = 128; s > 0; s >>= 1) { if (tid < s) red[tid] += red[tid + s]; __syncthreads(); }
    O[base + tid] = (h16)(e / red[0]);
}

__global__ void weighted_sum(const float* __restrict__ w,
                             const float* __restrict__ node32,
                             float* __restrict__ outrep, int off)
{
    const int b = blockIdx.x, d = threadIdx.x;   // 512 threads
    const float* nb = node32 + ((size_t)b * NNODE + TT) * DD + d;
    const float* wb = w + (size_t)b * TC;
    float acc = 0.0f;
    for (int c = 0; c < TC; ++c) acc += wb[c] * nb[(size_t)c * DD];
    outrep[(size_t)b * 1024 + off + d] = acc;
}

__global__ void final_linear(const float* __restrict__ rep, const float* __restrict__ W,
                             const float* __restrict__ bias, float* __restrict__ out)
{
    const int b = blockIdx.x, tid = threadIdx.x;   // 128 threads
    const int o = tid >> 5, lane = tid & 31;
    __shared__ float red[128];
    float acc = 0.0f;
    if (o < 3) {
        const float* rb = rep + (size_t)b * 1024;
        const float* wr = W + (size_t)o * 1024;
        for (int d = lane; d < 1024; d += 32) acc += rb[d] * wr[d];
    }
    red[tid] = acc; __syncthreads();
    for (int s = 16; s > 0; s >>= 1) { if (lane < s) red[tid] += red[tid + s]; __syncthreads(); }
    if (o < 3 && lane == 0) out[b * 3 + o] = red[tid] + bias[o];
}

// --------------------------------- launcher ---------------------------------

struct GruP { const float *wih, *whh, *bih, *bhh; };

extern "C" void kernel_launch(void* const* d_in, const int* in_sizes, int n_in,
                              void* d_out, int out_size, void* d_ws, size_t ws_size,
                              hipStream_t stream)
{
    (void)in_sizes; (void)n_in; (void)out_size;
    auto F = [&](int i) { return (const float*)d_in[i]; };

    const float* emb = F(9);
    GruP gp[4] = {
        {F(10), F(11), F(12), F(13)},   // task target
        {F(14), F(15), F(16), F(17)},   // task claim
        {F(18), F(19), F(20), F(21)},   // shared target
        {F(22), F(23), F(24), F(25)},   // shared claim
    };
    const int* idxs[4] = { (const int*)d_in[1], (const int*)d_in[3],
                           (const int*)d_in[2], (const int*)d_in[4] };
    const int  Ts[4]   = { TT, TC, TT, TC };

    char* wsp = (char*)d_ws;
    size_t cur = 0;
    auto alloc = [&](size_t bytes) -> void* {
        cur = (cur + 255) & ~(size_t)255;
        void* p = wsp + cur;
        cur += bytes;
        return p;
    };

    // ---- common workspace ----
    h16*   x_buf  = (h16*)  alloc((size_t)NB * TC * EP * 2);
    h16*   wih16  = (h16*)  alloc((size_t)G3 * EP * 2);
    h16*   hsT[2] = { (h16*)alloc((size_t)NB * TT * DD * 2),
                      (h16*)alloc((size_t)NB * TT * DD * 2) };
    h16*   hsC[2] = { (h16*)alloc((size_t)NB * TC * DD * 2),
                      (h16*)alloc((size_t)NB * TC * DD * 2) };
    float* attn_w = (float*)alloc((size_t)NB * TC * 4);
    h16*   node16 = (h16*)  alloc((size_t)NB * NNODE * DD * 2);
    float* node32 = (float*)alloc((size_t)NB * NNODE * DD * 4);
    float* corr32 = (float*)alloc((size_t)NB * NNODE * NNODE * 4);
    h16*   corr16 = (h16*)  alloc((size_t)NB * NNODE * NNODE * 2);
    h16*   trans16= (h16*)  alloc((size_t)NB * DD * NNODE * 2);
    h16*   gw16   = (h16*)  alloc((size_t)DD * DD * 2);
    float* frep   = (float*)alloc((size_t)NB * 1024 * 4);

    h16* hsb[4] = { hsT[0], hsC[0], hsT[1], hsC[1] };

    // ---- decide: fused (all 8 gi buffers live, one big GRU launch) vs
    //      sequential fallback (1 gi buffer, 8 GRU launches) ----
    const size_t gi_tgt_b = (size_t)NB * TT * G3 * 4;
    const size_t gi_clm_b = (size_t)NB * TC * G3 * 4;
    const size_t whh_b    = (size_t)G3 * HH * 2;
    const size_t need_fused = cur + 4 * (gi_tgt_b + gi_clm_b) + 8 * whh_b + 8192;
    const bool fused = (ws_size >= need_fused);

    if (fused) {
        float* giP[8];
        h16*   whhP[8];
        for (int gi_ = 0; gi_ < 4; ++gi_)
            for (int d = 0; d < 2; ++d) {
                giP[gi_ * 2 + d]  = (float*)alloc(Ts[gi_] == TT ? gi_tgt_b : gi_clm_b);
                whhP[gi_ * 2 + d] = (h16*)alloc(whh_b);
            }

        GruPack pk{};
        for (int gr = 0; gr < 4; ++gr) {
            const int rows = NB * Ts[gr];
            embed_gather<<<dim3((rows * EP + 255) / 256), 256, 0, stream>>>(
                x_buf, idxs[gr], emb, rows);
            for (int dir = 0; dir < 2; ++dir) {
                const int p = gr * 2 + dir;
                cast_pad<<<dim3((G3 * EP + 255) / 256), 256, 0, stream>>>(
                    wih16, gp[gr].wih + (size_t)dir * G3 * EE, G3, EE, EP);
                cast_pad<<<dim3((G3 * HH + 255) / 256), 256, 0, stream>>>(
                    whhP[p], gp[gr].whh + (size_t)dir * G3 * HH, G3, HH, HH);
                gemm16<0><<<dim3(rows / 32, G3 / 64, 1), 128, 0, stream>>>(
                    x_buf, 0, wih16, 0, giP[p], nullptr, 0,
                    gp[gr].bih + (size_t)dir * G3, rows, G3, EP, EP, EP);
                pk.p[p] = { giP[p], whhP[p], gp[gr].bhh + (size_t)dir * G3,
                            hsb[gr], Ts[gr], dir, dir ? HH : 0, 0 };
            }
        }
        // all 8 recurrent passes concurrently: 64 WGs, depth = max(T) = 200
        gru_seq_multi<<<dim3(NB / 16, 8), 512, 0, stream>>>(pk);
    } else {
        float* gi_buf = (float*)alloc(gi_clm_b);
        h16*   whhb   = (h16*)alloc(whh_b);
        for (int gr = 0; gr < 4; ++gr) {
            const int rows = NB * Ts[gr];
            embed_gather<<<dim3((rows * EP + 255) / 256), 256, 0, stream>>>(
                x_buf, idxs[gr], emb, rows);
            for (int dir = 0; dir < 2; ++dir) {
                cast_pad<<<dim3((G3 * EP + 255) / 256), 256, 0, stream>>>(
                    wih16, gp[gr].wih + (size_t)dir * G3 * EE, G3, EE, EP);
                cast_pad<<<dim3((G3 * HH + 255) / 256), 256, 0, stream>>>(
                    whhb, gp[gr].whh + (size_t)dir * G3 * HH, G3, HH, HH);
                gemm16<0><<<dim3(rows / 32, G3 / 64, 1), 128, 0, stream>>>(
                    x_buf, 0, wih16, 0, gi_buf, nullptr, 0,
                    gp[gr].bih + (size_t)dir * G3, rows, G3, EP, EP, EP);
                GruPack pk1{};
                pk1.p[0] = { gi_buf, whhb, gp[gr].bhh + (size_t)dir * G3,
                             hsb[gr], Ts[gr], dir, dir ? HH : 0, 0 };
                gru_seq_multi<<<dim3(NB / 16, 1), 512, 0, stream>>>(pk1);
            }
        }
    }

    // ---- attention + GCN + pooling per stance ----
    for (int st = 0; st < 2; ++st) {
        h16* tgt_hs = hsT[st];
        h16* clm_hs = hsC[st];

        attn_softmax<<<dim3(NB), 256, 0, stream>>>(tgt_hs, clm_hs, attn_w);
        node_init<<<dim3((unsigned)((size_t)NB * NNODE * DD / 256)), 256, 0, stream>>>(
            node16, node32, tgt_hs, clm_hs);

        const float* gw = F(st ? 27 : 26);
        for (int l = 0; l < 2; ++l) {
            cast_pad<<<dim3((DD * DD + 255) / 256), 256, 0, stream>>>(
                gw16, gw + (size_t)l * DD * DD, DD, DD, DD);
            // corr logits = node . node^T  (batched)
            gemm16<0><<<dim3(NNODE / 32, NNODE / 64, NB), 128, 0, stream>>>(
                node16, (size_t)NNODE * DD, node16, (size_t)NNODE * DD,
                corr32, nullptr, (size_t)NNODE * NNODE, nullptr,
                NNODE, NNODE, DD, DD, DD);
            softmax_corr<<<dim3(NNODE, NB), 256, 0, stream>>>(corr32, corr16);
            // trans^T (f16, transposed store):  trans = node @ W^T
            gemm16<1><<<dim3(NNODE / 32, DD / 64, NB), 128, 0, stream>>>(
                node16, (size_t)NNODE * DD, gw16, 0,
                nullptr, trans16, (size_t)DD * NNODE, nullptr,
                NNODE, DD, DD, DD, DD);
            // node = relu(node + corr @ trans)   (in place, dual f32+f16 store)
            gemm16<2><<<dim3(NNODE / 32, DD / 64, NB), 128, 0, stream>>>(
                corr16, (size_t)NNODE * NNODE, trans16, (size_t)DD * NNODE,
                node32, node16, (size_t)NNODE * DD, nullptr,
                NNODE, DD, NNODE, NNODE, NNODE);
        }
        weighted_sum<<<dim3(NB), 512, 0, stream>>>(attn_w, node32, frep, st * DD);
    }

    final_linear<<<dim3(NB), 128, 0, stream>>>(frep, F(28), F(29), (float*)d_out);
}
